// FourierNetwork_38345468019104
// MI455X (gfx1250) — matrix operand order repaired and checked by reading, not compiled, not run on hardware
//
#include <hip/hip_runtime.h>

// ---------------------------------------------------------------------------
// CDNA5 (gfx1250) implementation of the FourierNetwork reference.
// - Heavy matmuls (forward DFT, fused inverse-DFT + per-sample CxC mix, fc1)
//   run on v_wmma_f32_16x16x32_bf16.
// - Block-shared A operands staged in LDS via global_load_async_to_lds_b128
//   (ASYNCcnt); fc1 weight (32KB) fetched per-block by one TDM
//   tensor_load_to_lds descriptor (TENSORcnt).
// - Memory-bound w_spec expansion streams g2_w (537MB) with LDS-cached
//   activations + global_prefetch_b8.
// ---------------------------------------------------------------------------

typedef __attribute__((ext_vector_type(16))) __bf16 v16bf;
typedef __attribute__((ext_vector_type(8)))  float  v8f;
typedef __attribute__((ext_vector_type(4)))  unsigned int u32x4;
typedef __attribute__((ext_vector_type(8)))  unsigned int u32x8;

#define Bsz     32
#define Nn      4096
#define Cc      128
#define NMODES  32
#define NLAYERS 4
#define KINW    256
#define KMW     64
#define OUTW    (Cc * Cc * NMODES)   // 524288

__device__ __forceinline__ unsigned short f2bf(float f) {
  unsigned int u = __float_as_uint(f);
  u += 0x7FFFu + ((u >> 16) & 1u);   // round-to-nearest-even
  return (unsigned short)(u >> 16);
}
__device__ __forceinline__ float bf2f(unsigned short h) {
  return __uint_as_float(((unsigned int)h) << 16);
}
__device__ __forceinline__ float gelu_f(float v) {
  return 0.5f * v * (1.0f + erff(v * 0.70710678118654752440f));
}

union BfQuad { uint4 u; __bf16 b[8]; };

// Async global->LDS 16B copy (per enabled lane). LDS offset = low 32 bits of
// the flat shared pointer (ISA 10.2: flat LDS aperture addr[31:0]).
__device__ __forceinline__ void async_g2l_b128(void* lds_dst, const void* gsrc) {
  unsigned int loff = (unsigned int)(size_t)lds_dst;
  unsigned long long ga = (unsigned long long)(size_t)gsrc;
  asm volatile("global_load_async_to_lds_b128 %0, %1, off"
               :: "v"(loff), "v"(ga) : "memory");
}
__device__ __forceinline__ void wait_async0() {
  asm volatile("s_wait_asynccnt 0x0" ::: "memory");
}

// A fragment: 16x32 bf16 (M x K). Lanes 0-15 m=0..15 hold
// K 0..15; lanes 16-31 hold K 16..31.
__device__ __forceinline__ v16bf load_a_frag(const unsigned short* __restrict__ base,
                                             int lda, int kbase, int lane) {
  const int m = lane & 15, half = lane >> 4;
  const unsigned short* r = base + (size_t)m * lda + kbase + half * 16;
  BfQuad lo, hi;
  lo.u = *reinterpret_cast<const uint4*>(r);
  hi.u = *reinterpret_cast<const uint4*>(r + 8);
  v16bf a;
#pragma unroll
  for (int t = 0; t < 8; ++t) { a[t] = lo.b[t]; a[8 + t] = hi.b[t]; }
  return a;
}

// B fragment: 32x16 bf16 (K x N'). Lanes 0-15 (col n=0..15) hold K 0..15,
// lanes 16-31 hold K 16..31. Operand stored column-contiguous-in-K.
__device__ __forceinline__ v16bf load_b_frag(const unsigned short* __restrict__ base,
                                             int cstride, int kbase, int lane) {
  const int c = lane & 15, half = lane >> 4;
  const unsigned short* p = base + (size_t)c * cstride + kbase + half * 16;
  BfQuad q0, q1;
  q0.u = *reinterpret_cast<const uint4*>(p);
  q1.u = *reinterpret_cast<const uint4*>(p + 8);
  v16bf b;
#pragma unroll
  for (int t = 0; t < 8; ++t) { b[t] = q0.b[t]; b[8 + t] = q1.b[t]; }
  return b;
}

// ---------------------------------------------------------------------------
// Basis tables. FB[j][n] (j<32: cos, j>=32: -sin) feeds the forward DFT as a
// K-contiguous B operand. IBT[n][j] (cos | sin) feeds the inverse DFT.
// Exact phase reduction: (m*n) mod N.
// ---------------------------------------------------------------------------
__global__ void basis_kernel(unsigned short* __restrict__ FB,
                             unsigned short* __restrict__ IBT) {
  int idx = blockIdx.x * 256 + threadIdx.x;      // 64*4096 threads
  int j = idx >> 12, n = idx & (Nn - 1);
  int mm = j & 31;
  int ph = (mm * n) & (Nn - 1);
  float th = (float)ph * 1.5339807878856412e-3f; // 2*pi/4096
  float cv = cosf(th), sv = sinf(th);
  FB[(size_t)j * Nn + n]  = f2bf(j < 32 ? cv : -sv);
  IBT[(size_t)n * 64 + j] = f2bf(j < 32 ? cv : sv);
}

__global__ void fc1t_kernel(const float* __restrict__ fc1w,
                            unsigned short* __restrict__ FC1T) {
  int idx = blockIdx.x * 256 + threadIdx.x;      // 16384
  int j = idx >> 7, c = idx & 127;
  FC1T[(size_t)j * Cc + c] = f2bf(fc1w[(size_t)c * 128 + j]);
}

// fc0: [B,N,3]@[3,C]+b -> x stored [B][C][N] (XB) and [B][N][C] (XT), bf16.
__global__ void fc0_kernel(const float* __restrict__ X,
                           const float* __restrict__ W,
                           const float* __restrict__ bias,
                           unsigned short* __restrict__ XB,
                           unsigned short* __restrict__ XT) {
  int bn = blockIdx.x;                  // b*N + n
  int c  = threadIdx.x;                 // 0..127
  int b  = bn >> 12, n = bn & (Nn - 1);
  const float* xr = X + (size_t)bn * 3;
  float v = xr[0] * W[c] + xr[1] * W[128 + c] + xr[2] * W[256 + c] + bias[c];
  unsigned short h = f2bf(v);
  XB[((size_t)(b * Cc + c)) * Nn + n] = h;
  XT[((size_t)b * Nn + n) * Cc + c]   = h;
}

// gelu(y @ W + bias): [32,256]@[256,64]. One block per sample, 64 threads.
__global__ void dense64_kernel(const float* __restrict__ y,
                               const float* __restrict__ W,
                               const float* __restrict__ bias,
                               float* __restrict__ out) {
  int b = blockIdx.x, t = threadIdx.x;
  const float* yb = y + (size_t)b * KINW;
  float acc = bias[t];
#pragma unroll 4
  for (int k = 0; k < KINW; ++k) acc += yb[k] * W[(size_t)k * KMW + t];
  out[(size_t)b * KMW + t] = gelu_f(acc);
}

// h[B,64] @ W[64,M] + bias -> out[B][M]. Memory bound on W: each thread owns
// one column j, keeps 32 accumulators (all samples), h cached in LDS.
__global__ void expand_kernel(const float* __restrict__ h,
                              const float* __restrict__ W,
                              const float* __restrict__ bias,
                              float* __restrict__ outf,
                              unsigned short* __restrict__ outh,
                              int M, int as_bf16) {
  __shared__ float sh[Bsz * KMW];      // 8 KB
  for (int t = threadIdx.x; t < Bsz * KMW; t += 256) sh[t] = h[t];
  __syncthreads();
  int j = blockIdx.x * 256 + threadIdx.x;
  float acc[Bsz];
  float bj = bias[j];
#pragma unroll
  for (int b = 0; b < Bsz; ++b) acc[b] = bj;
  for (int k = 0; k < KMW; ++k) {
    if (k + 4 < KMW)
      __builtin_prefetch((const void*)(W + (size_t)(k + 4) * M + j), 0, 0);
    float w = W[(size_t)k * M + j];
#pragma unroll
    for (int b = 0; b < Bsz; ++b) acc[b] += sh[b * KMW + k] * w;
  }
  if (as_bf16) {
#pragma unroll
    for (int b = 0; b < Bsz; ++b) outh[(size_t)b * M + j] = f2bf(acc[b]);
  } else {
#pragma unroll
    for (int b = 0; b < Bsz; ++b) outf[(size_t)b * M + j] = acc[b];
  }
}

// Forward DFT (WMMA): XFT[4096][64] = X_bf16[4096][4096] @ F[4096][64].
__global__ void fwd_dft_kernel(const unsigned short* __restrict__ X,
                               const unsigned short* __restrict__ FB,
                               float* __restrict__ XFT) {
  int wave = (blockIdx.x * blockDim.x + threadIdx.x) >> 5;  // 1024 waves
  int lane = threadIdx.x & 31;
  int rt = wave >> 2, jt = wave & 3;
  const unsigned short* A = X + (size_t)rt * 16 * Nn;
  const unsigned short* Bp = FB + (size_t)jt * 16 * Nn;
  v8f c = {0.f, 0.f, 0.f, 0.f, 0.f, 0.f, 0.f, 0.f};
  for (int k = 0; k < Nn; k += 32) {
    v16bf a = load_a_frag(A, Nn, k, lane);
    v16bf b = load_b_frag(Bp, Nn, k, lane);
    c = __builtin_amdgcn_wmma_f32_16x16x32_bf16(false, a, false, b,
                                                (short)0, c, false, false);
  }
  int col = lane & 15, half = lane >> 4;
#pragma unroll
  for (int v = 0; v < 8; ++v)
    XFT[(size_t)(rt * 16 + v + half * 8) * 64 + jt * 16 + col] = c[v];
}

// Mode mixing: om[b,o,m] = sum_i xft[b,i,m]*w_spec[b,i,o,m] (real weights so
// Re/Im independent). Bakes irfft scale (2-d_{m0})/N and -Im sign into om.
__global__ void mix_kernel(const float* __restrict__ XFT,
                           const float* __restrict__ WS,
                           unsigned short* __restrict__ OM) {
  int idx = blockIdx.x * 256 + threadIdx.x;   // B*C*MODES = 131072
  int m = idx & 31, o = (idx >> 5) & 127, b = idx >> 12;
  float cm = (m == 0 ? 1.0f : 2.0f) * (1.0f / (float)Nn);
  const float* xf = XFT + (size_t)(b * Cc) * 64;
  const float* wp = WS + (((size_t)b * Cc) * Cc + o) * NMODES + m;
  float ar = 0.f, ai = 0.f;
#pragma unroll 4
  for (int i = 0; i < Cc; ++i) {
    float w = wp[(size_t)i * Cc * NMODES];
    ar += xf[i * 64 + m] * w;
    ai += xf[i * 64 + 32 + m] * w;
  }
  size_t r = (size_t)(b * Cc + o) * 64;
  OM[r + m]      = f2bf(cm * ar);    // cos coeff
  OM[r + 32 + m] = f2bf(-cm * ai);   // sin coeff (Im stored as -sum x sin)
}

// Fused per-layer output (WMMA): x_new = irDFT(om) + w_lin[b] @ x_old, gelu.
// Each block = 8 waves = one 16-row tile x 8 column tiles, so the A operands
// (om 16x64, w_lin 16x128 -> 6KB) are staged once per block in LDS via
// global_load_async_to_lds_b128 and reused by all 8 waves.
__global__ void layer_out_kernel(const unsigned short* __restrict__ OM,
                                 const unsigned short* __restrict__ IBT,
                                 const unsigned short* __restrict__ WL,
                                 const unsigned short* __restrict__ XT,
                                 unsigned short* __restrict__ XBn,
                                 unsigned short* __restrict__ XTn,
                                 int apply_gelu) {
  __shared__ __align__(16) unsigned short sA1[16 * 64];    // om rows (2KB)
  __shared__ __align__(16) unsigned short sA2[16 * Cc];    // w_lin rows (4KB)
  int tid = threadIdx.x;
  int wave = (blockIdx.x * blockDim.x + tid) >> 5;  // 65536 waves
  int lane = tid & 31;
  int rt = wave >> 8, nt = wave & 255;
  int b = rt >> 3;                                  // 8 row-tiles per sample

  {
    const unsigned short* gA1 = OM + (size_t)rt * 16 * 64;  // 2KB contiguous
    const unsigned short* gA2 = WL + (size_t)rt * 16 * Cc;  // 4KB contiguous
    if (tid < 128) async_g2l_b128(&sA1[tid * 8], gA1 + tid * 8);
    async_g2l_b128(&sA2[tid * 8], gA2 + tid * 8);
    wait_async0();
    __syncthreads();
  }

  v8f c = {0.f, 0.f, 0.f, 0.f, 0.f, 0.f, 0.f, 0.f};
  // Phase 1: inverse DFT, K=64 over mode basis.
  const unsigned short* B1 = IBT + (size_t)nt * 16 * 64;
#pragma unroll
  for (int k = 0; k < 64; k += 32) {
    v16bf a = load_a_frag(sA1, 64, k, lane);
    v16bf bb = load_b_frag(B1, 64, k, lane);
    c = __builtin_amdgcn_wmma_f32_16x16x32_bf16(false, a, false, bb,
                                                (short)0, c, false, false);
  }
  // Phase 2: per-sample CxC mix, K=128 over channels.
  const unsigned short* B2 = XT + ((size_t)b * Nn + nt * 16) * Cc;
#pragma unroll
  for (int k = 0; k < Cc; k += 32) {
    v16bf a = load_a_frag(sA2, Cc, k, lane);
    v16bf bb = load_b_frag(B2, Cc, k, lane);
    c = __builtin_amdgcn_wmma_f32_16x16x32_bf16(false, a, false, bb,
                                                (short)0, c, false, false);
  }
  int col = lane & 15, half = lane >> 4;
  int n = nt * 16 + col;
#pragma unroll
  for (int v = 0; v < 8; ++v) {
    int g = rt * 16 + v + half * 8;              // global row b*C+o
    float val = c[v];
    if (apply_gelu) val = gelu_f(val);
    unsigned short hbits = f2bf(val);
    XBn[(size_t)g * Nn + n] = hbits;
    XTn[((size_t)b * Nn + n) * Cc + (g & 127)] = hbits;
  }
}

// fc1 (WMMA): H = gelu(XT[131072][128] @ fc1_w + b). The full 32KB fc1
// weight is DMA'd into LDS once per block by a single TDM descriptor
// (tensor_load_to_lds, TENSORcnt); A rows staged via async-to-LDS.
__global__ void fc1_kernel(const unsigned short* __restrict__ XT,
                           const unsigned short* __restrict__ FC1T,
                           const float* __restrict__ bias,
                           unsigned short* __restrict__ H) {
  __shared__ __align__(16) unsigned short sW[128 * Cc];    // 32KB weight
  __shared__ __align__(16) unsigned short sA[16 * Cc];     // 4KB A rows
  int tid = threadIdx.x;
  int wave = (blockIdx.x * blockDim.x + tid) >> 5;  // 65536 waves
  int lane = tid & 31;
  int rt = wave >> 3, jt = wave & 7;

  if ((tid >> 5) == 0) {
    // D# group 0: count=1 | lds_addr | global_addr[56:0] | type=2
    unsigned long long ga = (unsigned long long)(size_t)FC1T;
    u32x4 g0;
    g0[0] = 1u;                                    // count=1, user mode
    g0[1] = (unsigned int)(size_t)&sW[0];          // lds_addr (bytes)
    g0[2] = (unsigned int)(ga & 0xFFFFFFFFu);      // global_addr[31:0]
    g0[3] = ((unsigned int)(ga >> 32) & 0x01FFFFFFu) | (2u << 30); // +type=2
    // D# group 1: data_size=2B; tensor 128x128; tile 128x128; dim0 stride 128
    u32x8 g1;
    g1[0] = 1u << 16;          // data_size=1 (2 bytes), no multicast/pad
    g1[1] = 128u << 16;        // tensor_dim0 = 128    (bits 79:48)
    g1[2] = 128u << 16;        // tensor_dim1 = 128    (bits 111:80)
    g1[3] = 128u << 16;        // tile_dim0   = 128    (bits 127:112)
    g1[4] = 128u;              // tile_dim1   = 128    (bits 143:128)
    g1[5] = 128u;              // tensor_dim0_stride = 128 (bits 207:160)
    g1[6] = 0u;
    g1[7] = 0u;
    asm volatile("tensor_load_to_lds %0, %1" :: "s"(g0), "s"(g1) : "memory");
    __builtin_amdgcn_s_wait_tensorcnt(0);
  }
  {
    const unsigned short* gA = XT + (size_t)rt * 16 * Cc;   // 4KB contiguous
    async_g2l_b128(&sA[tid * 8], gA + tid * 8);
    wait_async0();
  }
  __syncthreads();

  v8f c = {0.f, 0.f, 0.f, 0.f, 0.f, 0.f, 0.f, 0.f};
#pragma unroll
  for (int k = 0; k < Cc; k += 32) {
    v16bf a = load_a_frag(sA, Cc, k, lane);
    v16bf bb = load_b_frag(sW + (size_t)jt * 16 * Cc, Cc, k, lane);
    c = __builtin_amdgcn_wmma_f32_16x16x32_bf16(false, a, false, bb,
                                                (short)0, c, false, false);
  }
  int col = lane & 15, half = lane >> 4;
  int j = jt * 16 + col;
  float bj = bias[j];
#pragma unroll
  for (int v = 0; v < 8; ++v) {
    int row = rt * 16 + v + half * 8;
    H[(size_t)row * 128 + j] = f2bf(gelu_f(c[v] + bj));
  }
}

// fc2: out[row] = (sum_j H[row][j]*fc2_w[j] + fc2_b) * weightx
__global__ void fc2_kernel(const unsigned short* __restrict__ H,
                           const float* __restrict__ w,
                           const float* __restrict__ bias,
                           const float* __restrict__ wx,
                           float* __restrict__ out) {
  int row = blockIdx.x * 256 + threadIdx.x;    // 131072
  const unsigned short* hr = H + (size_t)row * 128;
  float s = bias[0];
#pragma unroll 4
  for (int j = 0; j < 128; ++j) s += bf2f(hr[j]) * w[j];
  out[row] = s * wx[0];
}

// ---------------------------------------------------------------------------
extern "C" void kernel_launch(void* const* d_in, const int* in_sizes, int n_in,
                              void* d_out, int out_size, void* d_ws, size_t ws_size,
                              hipStream_t stream) {
  (void)in_sizes; (void)n_in; (void)out_size; (void)ws_size;
  const float* x    = (const float*)d_in[0];
  const float* y    = (const float*)d_in[1];
  const float* fc0w = (const float*)d_in[2];
  const float* fc0b = (const float*)d_in[3];
  const float* g1w  = (const float*)d_in[4];
  const float* g1b  = (const float*)d_in[5];
  const float* g2w  = (const float*)d_in[6];
  const float* g2b  = (const float*)d_in[7];
  const float* l1w  = (const float*)d_in[8];
  const float* l1b  = (const float*)d_in[9];
  const float* l2w  = (const float*)d_in[10];
  const float* l2b  = (const float*)d_in[11];
  const float* fc1w = (const float*)d_in[12];
  const float* fc1b = (const float*)d_in[13];
  const float* fc2w = (const float*)d_in[14];
  const float* fc2b = (const float*)d_in[15];
  const float* wx   = (const float*)d_in[16];

  char* ws = (char*)d_ws;
  size_t off = 0;
  auto take = [&](size_t bytes) -> char* {
    char* p = ws + off;
    off += (bytes + 255) & ~(size_t)255;
    return p;
  };
  unsigned short* XB[2]; unsigned short* XT[2];
  XB[0] = (unsigned short*)take((size_t)Bsz * Cc * Nn * 2);
  XT[0] = (unsigned short*)take((size_t)Bsz * Nn * Cc * 2);
  XB[1] = (unsigned short*)take((size_t)Bsz * Cc * Nn * 2);
  XT[1] = (unsigned short*)take((size_t)Bsz * Nn * Cc * 2);
  float*          WSPEC = (float*)take((size_t)Bsz * OUTW * 4);
  unsigned short* WLIN  = (unsigned short*)take((size_t)Bsz * Cc * Cc * 2);
  float*          HG    = (float*)take((size_t)Bsz * KMW * 4);
  float*          HL    = (float*)take((size_t)Bsz * KMW * 4);
  unsigned short* FB    = (unsigned short*)take((size_t)64 * Nn * 2);
  unsigned short* IBT   = (unsigned short*)take((size_t)Nn * 64 * 2);
  float*          XFT   = (float*)take((size_t)Bsz * Cc * 64 * 4);
  unsigned short* OM    = (unsigned short*)take((size_t)Bsz * Cc * 64 * 2);
  unsigned short* FC1T  = (unsigned short*)take((size_t)128 * Cc * 2);
  unsigned short* H     = (unsigned short*)WSPEC;   // reuse (67MB > 33.5MB)

  basis_kernel<<<1024, 256, 0, stream>>>(FB, IBT);
  fc1t_kernel<<<64, 256, 0, stream>>>(fc1w, FC1T);
  fc0_kernel<<<Bsz * Nn, 128, 0, stream>>>(x, fc0w, fc0b, XB[0], XT[0]);

  int cur = 0;
  for (int i = 0; i < NLAYERS; ++i) {
    dense64_kernel<<<Bsz, 64, 0, stream>>>(
        y, g1w + (size_t)i * KINW * KMW, g1b + (size_t)i * KMW, HG);
    dense64_kernel<<<Bsz, 64, 0, stream>>>(
        y, l1w + (size_t)i * KINW * KMW, l1b + (size_t)i * KMW, HL);
    expand_kernel<<<OUTW / 256, 256, 0, stream>>>(
        HG, g2w + (size_t)i * KMW * OUTW, g2b + (size_t)i * OUTW,
        WSPEC, nullptr, OUTW, 0);
    expand_kernel<<<(Cc * Cc) / 256, 256, 0, stream>>>(
        HL, l2w + (size_t)i * KMW * Cc * Cc, l2b + (size_t)i * Cc * Cc,
        nullptr, WLIN, Cc * Cc, 1);
    fwd_dft_kernel<<<128, 256, 0, stream>>>(XB[cur], FB, XFT);
    mix_kernel<<<512, 256, 0, stream>>>(XFT, WSPEC, OM);
    layer_out_kernel<<<8192, 256, 0, stream>>>(
        OM, IBT, WLIN, XT[cur], XB[cur ^ 1], XT[cur ^ 1],
        (i < NLAYERS - 1) ? 1 : 0);
    cur ^= 1;
  }
  fc1_kernel<<<8192, 256, 0, stream>>>(XT[cur], FC1T, fc1b, H);
  fc2_kernel<<<512, 256, 0, stream>>>(H, fc2w, fc2b, wx, (float*)d_out);
}
